// VQKD_58033598104214
// MI455X (gfx1250) — compile-verified, matrix operand order checked by reading, not run on hardware
//
#include <hip/hip_runtime.h>

typedef __attribute__((ext_vector_type(16))) __bf16 v16bf;
typedef __attribute__((ext_vector_type(8)))  float  v8f;
typedef __attribute__((ext_vector_type(4)))  unsigned int v4u;
typedef __attribute__((ext_vector_type(8)))  int    v8i_t;
typedef __attribute__((ext_vector_type(4)))  int    v4i_t;

#define M_TOTAL   32768
#define D_ENC     768
#define D_CODE    32
#define K_CODES   8192
#define M_BLK     32
#define ZQ_ELEMS  (M_TOTAL * D_CODE)   // 1048576 floats

// ---- workspace byte offsets ----
#define WS_EMB_F32   0          // 8192*32*4  = 1,048,576
#define WS_EMB_BF16  1048576    // 8192*32*2  =   524,288
#define WS_W1T       1572864    // 768*768*2  = 1,179,648
#define WS_W2T       2752512    // 32*768*2   =    49,152
#define WS_PART      2801664    // 1024*4     =     4,096

// ---- LDS element strides (bank-conflict padded; TDM pad reproduces 40) ----
#define A_STRIDE  776   // 768 + 8 bf16
#define H_STRIDE  776
#define W_STRIDE  40    // 32 + 8 bf16 (TDM: 16 dwords data + 4 dwords pad)
#define E_STRIDE  40
#define ZB_STRIDE 40
#define Z_STRIDE  33    // 32 + 1 f32

// ---- LDS byte offsets (all 16B aligned) ----
#define LDS_A     0        // 32*776*2  = 49,664
#define LDS_H     49664    // 32*776*2  = 49,664
#define LDS_W     99328    // 2 buf x 128*40*2 = 2 x 10,240
#define LDS_E     119808   // 2 buf x 128*40*2 = 2 x 10,240
#define LDS_Z     140288   // 32*33*4   =  4,224
#define LDS_ZB    144512   // 32*40*2   =  2,560
#define LDS_RV    147072   // 4*32*4    =    512
#define LDS_RI    147584   // 4*32*4    =    512
#define LDS_IDX   148096   // 32*4      =    128
#define LDS_PART  148224   // 8*4       =     32
#define LDS_BYTES 148480   // < 160KB -> 2 workgroups / WGP

#define WBUF_BYTES 10240
#define EBUF_BYTES 10240

__device__ __forceinline__ unsigned short f2bf(float f) {
  unsigned u = __float_as_uint(f);
  unsigned r = (u + 0x7FFFu + ((u >> 16) & 1u)) >> 16;  // round-to-nearest-even
  return (unsigned short)r;
}

// Fragment = two contiguous 16B runs (k = 8h..8h+7 and 16+8h..16+8h+7) per lane.
__device__ __forceinline__ v16bf frag16(const unsigned short* p) {
  union { uint4 q[2]; v16bf v; } u;
  u.q[0] = *reinterpret_cast<const uint4*>(p);
  u.q[1] = *reinterpret_cast<const uint4*>(p + 16);
  return u.v;
}

__device__ __forceinline__ v8f wmma_bf16(v16bf a, v16bf b, v8f c) {
  return __builtin_amdgcn_wmma_f32_16x16x32_bf16(false, a, false, b, (short)0, c,
                                                 false, false);
}

// ---- Tensor Data Mover: 2-D bf16 tile (tile_d0 elems/row x tile_d1 rows) ----
// LDS dest gets +16B pad after every 64B row => element stride 40.
__device__ __forceinline__ void tdm_load_2d(unsigned lds_byte_off,
                                            const void* gptr,
                                            unsigned tile_d0,          // elements (2B)
                                            unsigned tile_d1,          // rows
                                            unsigned row_stride_elems, // tensor dim0 stride
                                            unsigned tensor_d1) {     // rows in tensor
  unsigned long long ga = (unsigned long long)gptr;
  v4u g0 = { 1u,                                    // count=1, user mode
             lds_byte_off,                          // lds_addr
             (unsigned)(ga & 0xFFFFFFFFu),          // global_addr[31:0]
             ((unsigned)((ga >> 32) & 0x01FFFFFFu)) // global_addr[56:32]
                 | 0x80000000u };                   // type=2 ("image")
  // data_size=1 (2B); pad_enable; pad_interval=3 (16 dwords); pad_amount=3 (4 dwords)
  unsigned d0 = (1u << 16) | (1u << 20) | (3u << 22) | (3u << 25);
  unsigned td0 = row_stride_elems;                  // tensor_dim0 (in-bounds use)
  v8i_t g1 = { (int)d0,
               (int)((td0 & 0xFFFFu) << 16),                                  // dim0 lo
               (int)(((td0 >> 16) & 0xFFFFu) | ((tensor_d1 & 0xFFFFu) << 16)),// dim0 hi | dim1 lo
               (int)(((tensor_d1 >> 16) & 0xFFFFu) | (tile_d0 << 16)),        // dim1 hi | tile0
               (int)(tile_d1 & 0xFFFFu),                                      // tile1 | tile2=0
               (int)row_stride_elems,                                         // dim0_stride lo
               0, 0 };
  v4i_t gz = {0, 0, 0, 0};
#if __clang_major__ >= 23
  v8i_t gz8 = {0, 0, 0, 0, 0, 0, 0, 0};
  __builtin_amdgcn_tensor_load_to_lds(g0, g1, gz, gz, gz8, 0);
#else
  __builtin_amdgcn_tensor_load_to_lds(g0, g1, gz, gz, 0);
#endif
}

// ---------- prep: transpose + f32->bf16 (out[n*K + k] = in[k*N + n]) ----------
__global__ __launch_bounds__(256) void k_transpose_bf16(const float* __restrict__ in,
                                                        unsigned short* __restrict__ out,
                                                        int K, int N) {
  int i = blockIdx.x * 256 + threadIdx.x;
  if (i >= K * N) return;
  int n = i / K, k = i % K;
  out[n * K + k] = f2bf(in[k * N + n]);
}

// ---------- prep: l2-normalize codebook rows -> f32 + bf16 copies ----------
__global__ __launch_bounds__(256) void k_norm_codebook(const float* __restrict__ cb,
                                                       float* __restrict__ emb_f32,
                                                       unsigned short* __restrict__ emb_bf16) {
  int w = threadIdx.x >> 5, lane = threadIdx.x & 31;
  int row = blockIdx.x * 8 + w;
  float v = cb[row * 32 + lane];
  float s = v * v;
  #pragma unroll
  for (int m = 16; m >= 1; m >>= 1) s += __shfl_xor(s, m, 32);
  float inv = 1.0f / fmaxf(sqrtf(s), 1e-12f);
  float nv = v * inv;
  emb_f32[row * 32 + lane] = nv;
  emb_bf16[row * 32 + lane] = f2bf(nv);
}

// ---------- fused VQKD: GEMM1(tanh) -> GEMM2 -> l2norm -> argmax -> gather/loss ----------
__global__ __launch_bounds__(256) void k_vqkd_fused(
    const float* __restrict__ features,
    const float* __restrict__ b1,
    const float* __restrict__ b2,
    const unsigned short* __restrict__ W1T,      // [768 n][768 k] bf16
    const unsigned short* __restrict__ W2T,      // [32 n][768 k] bf16
    const unsigned short* __restrict__ emb_bf16, // [8192][32]
    const float* __restrict__ emb_f32,           // [8192][32]
    float* __restrict__ out,                     // z_q | loss | idx(int)
    float* __restrict__ partials) {
  extern __shared__ unsigned char smem[];
  unsigned short* A_lds  = (unsigned short*)(smem + LDS_A);
  unsigned short* h_lds  = (unsigned short*)(smem + LDS_H);
  unsigned short* w_lds  = (unsigned short*)(smem + LDS_W);
  unsigned short* e_lds  = (unsigned short*)(smem + LDS_E);
  float*          z_lds  = (float*)(smem + LDS_Z);
  unsigned short* zb_lds = (unsigned short*)(smem + LDS_ZB);
  float*          rv_lds = (float*)(smem + LDS_RV);
  int*            ri_lds = (int*)(smem + LDS_RI);
  int*            idx_lds = (int*)(smem + LDS_IDX);
  float*          pl_lds  = (float*)(smem + LDS_PART);

  const int tid  = threadIdx.x;
  const int w    = tid >> 5;
  const int lane = tid & 31;
  const int nn   = lane & 15;   // row/col within 16-wide fragment
  const int hh   = lane >> 4;   // half-wave selector for K runs
  const int m0   = blockIdx.x * M_BLK;

  // ---- stage feature block [32][768] f32 -> bf16 in LDS (read once from HBM) ----
  #pragma unroll
  for (int i = 0; i < 24; ++i) {
    int li = tid + i * 256;
    int m  = li / 192;
    int k  = (li % 192) * 4;
    const float4 f = *reinterpret_cast<const float4*>(
        &features[(size_t)(m0 + m) * D_ENC + k]);
    uint2 u;
    u.x = (unsigned)f2bf(f.x) | ((unsigned)f2bf(f.y) << 16);
    u.y = (unsigned)f2bf(f.z) | ((unsigned)f2bf(f.w) << 16);
    *reinterpret_cast<uint2*>(&A_lds[m * A_STRIDE + k]) = u;
  }

  // ---- GEMM1: h = tanh(A @ W1 + b1), h kept in LDS ----
  // W1T tiles [128 n][32 k] arrive via double-buffered TDM DMA issued by wave 0.
  if (w == 0) tdm_load_2d(LDS_W, &W1T[0], 32, 128, D_ENC, D_ENC);  // stage 0
  __syncthreads();  // A_lds visible

  const int mtile = w & 1;      // 2 M-strips of 16
  const int npair = w >> 1;     // 4 pairs of N-tiles within 128-col block
  int sidx = 0;                 // linear stage index over (n0,k0)
  for (int n0 = 0; n0 < D_ENC; n0 += 128) {
    v8f acc0 = {0, 0, 0, 0, 0, 0, 0, 0};
    v8f acc1 = {0, 0, 0, 0, 0, 0, 0, 0};
    const int nA = n0 + npair * 32 + nn;
    const int nB = nA + 16;
    const float biasA = b1[nA];
    const float biasB = b1[nB];
    for (int k0 = 0; k0 < D_ENC; k0 += 32, ++sidx) {
      __syncthreads();                      // everyone done with buf (sidx-1)&1
      if (w == 0) {
        int sn = sidx + 1;
        if (sn < 144) {                     // 6 n-blocks * 24 k-steps
          int nn0 = (sn / 24) * 128, nk0 = (sn % 24) * 32;
          tdm_load_2d(LDS_W + (unsigned)(sn & 1) * WBUF_BYTES,
                      &W1T[nn0 * D_ENC + nk0], 32, 128, D_ENC, D_ENC);
          __builtin_amdgcn_s_wait_tensorcnt(1);  // current tile landed
        } else {
          __builtin_amdgcn_s_wait_tensorcnt(0);
        }
      }
      __syncthreads();                      // current buf ready for all waves
      const unsigned short* wb = w_lds + (sidx & 1) * (128 * W_STRIDE);
      v16bf af  = frag16(&A_lds[(mtile * 16 + nn) * A_STRIDE + k0 + 8 * hh]);
      v16bf bf0 = frag16(&wb[(npair * 32 + nn) * W_STRIDE + 8 * hh]);
      v16bf bf1 = frag16(&wb[(npair * 32 + 16 + nn) * W_STRIDE + 8 * hh]);
      acc0 = wmma_bf16(af, bf0, acc0);
      acc1 = wmma_bf16(af, bf1, acc1);
    }
    #pragma unroll
    for (int r = 0; r < 8; ++r) {           // C layout: row = r + 8*hh, col = nn
      int mrow = mtile * 16 + r + 8 * hh;
      h_lds[mrow * H_STRIDE + nA] = f2bf(tanhf(acc0[r] + biasA));
      h_lds[mrow * H_STRIDE + nB] = f2bf(tanhf(acc1[r] + biasB));
    }
  }
  __syncthreads();

  // prefetch codebook chunk 0 via TDM; overlaps GEMM2 + normalize
  if (w == 0) tdm_load_2d(LDS_E, emb_bf16, 32, 128, 32, K_CODES);

  // ---- GEMM2: z = h @ W2 + b2 (waves 0..3; B frags straight from L2-hot W2T) ----
  if (w < 4) {
    const int mt2 = w & 1;
    const int nt2 = w >> 1;
    v8f acc = {0, 0, 0, 0, 0, 0, 0, 0};
    for (int k0 = 0; k0 < D_ENC; k0 += 32) {
      v16bf af = frag16(&h_lds[(mt2 * 16 + nn) * H_STRIDE + k0 + 8 * hh]);
      v16bf bf = frag16(&W2T[(nt2 * 16 + nn) * D_ENC + k0 + 8 * hh]);
      acc = wmma_bf16(af, bf, acc);
    }
    const int c = nt2 * 16 + nn;
    const float bias = b2[c];
    #pragma unroll
    for (int r = 0; r < 8; ++r)
      z_lds[(mt2 * 16 + r + 8 * hh) * Z_STRIDE + c] = acc[r] + bias;
  }
  __syncthreads();

  // ---- l2-normalize z rows; keep f32 (loss) + bf16 (scoring) ----
  if (tid < 32) {
    float s = 0.f;
    #pragma unroll
    for (int c = 0; c < 32; ++c) { float v = z_lds[tid * Z_STRIDE + c]; s += v * v; }
    float inv = 1.0f / fmaxf(sqrtf(s), 1e-12f);
    #pragma unroll
    for (int c = 0; c < 32; ++c) {
      float nv = z_lds[tid * Z_STRIDE + c] * inv;
      z_lds[tid * Z_STRIDE + c] = nv;
      zb_lds[tid * ZB_STRIDE + c] = f2bf(nv);
    }
  }
  __syncthreads();

  // ---- scoring: argmax_j z . emb_j ; one WMMA per 16x16 score tile (K=32) ----
  // 64 chunks of 128 codes, double-buffered TDM prefetch of chunk c+1 during chunk c.
  const int strip = w & 1;   // M-strip
  const int jsub  = w >> 1;  // quarter of each 128-code chunk
  v16bf zfrag = frag16(&zb_lds[(strip * 16 + nn) * ZB_STRIDE + 8 * hh]);
  float best[8];
  int   bidx[8];
  #pragma unroll
  for (int r = 0; r < 8; ++r) { best[r] = -3.402823466e38f; bidx[r] = 0; }

  for (int c = 0; c < 64; ++c) {
    __syncthreads();                        // buf (c+1)&1 free for overwrite
    if (w == 0) {
      if (c + 1 < 64) {
        tdm_load_2d(LDS_E + (unsigned)((c + 1) & 1) * EBUF_BYTES,
                    &emb_bf16[(c + 1) * 128 * 32], 32, 128, 32, K_CODES);
        __builtin_amdgcn_s_wait_tensorcnt(1);
      } else {
        __builtin_amdgcn_s_wait_tensorcnt(0);
      }
    }
    __syncthreads();                        // chunk c resident
    const unsigned short* eb = e_lds + (c & 1) * (128 * E_STRIDE);
    #pragma unroll
    for (int t2 = 0; t2 < 2; ++t2) {
      int jl = jsub * 32 + t2 * 16;
      v16bf ef = frag16(&eb[(jl + nn) * E_STRIDE + 8 * hh]);
      v8f zero = {0, 0, 0, 0, 0, 0, 0, 0};
      v8f sc = wmma_bf16(zfrag, ef, zero);
      int jg = c * 128 + jl + nn;
      #pragma unroll
      for (int r = 0; r < 8; ++r) {
        bool g = sc[r] > best[r];
        best[r] = g ? sc[r] : best[r];
        bidx[r] = g ? jg : bidx[r];
      }
    }
  }
  // reduce across the 16 lanes holding one row (ties -> lowest index = argmin-first)
  #pragma unroll
  for (int mask = 1; mask <= 8; mask <<= 1) {
    #pragma unroll
    for (int r = 0; r < 8; ++r) {
      float ov = __shfl_xor(best[r], mask, 32);
      int   oi = __shfl_xor(bidx[r], mask, 32);
      bool g = (ov > best[r]) || (ov == best[r] && oi < bidx[r]);
      best[r] = g ? ov : best[r];
      bidx[r] = g ? oi : bidx[r];
    }
  }
  if (nn == 0) {
    #pragma unroll
    for (int r = 0; r < 8; ++r) {
      int row = strip * 16 + r + 8 * hh;
      rv_lds[jsub * 32 + row] = best[r];
      ri_lds[jsub * 32 + row] = bidx[r];
    }
  }
  __syncthreads();
  if (tid < 32) {                           // combine 4 j-partials per row
    float bv = rv_lds[tid];
    int   bi = ri_lds[tid];
    #pragma unroll
    for (int js = 1; js < 4; ++js) {
      float v = rv_lds[js * 32 + tid];
      int  i2 = ri_lds[js * 32 + tid];
      bool g = (v > bv) || (v == bv && i2 < bi);
      bv = g ? v : bv;
      bi = g ? i2 : bi;
    }
    idx_lds[tid] = bi;
    reinterpret_cast<int*>(out)[ZQ_ELEMS + 1 + m0 + tid] = bi;  // idx output
  }
  __syncthreads();

  // ---- gather z_q = emb[idx], store, and commitment-loss partial ----
  {
    int row = tid >> 3;
    int c4  = (tid & 7) * 4;
    int id  = idx_lds[row];
    float4 e = *reinterpret_cast<const float4*>(&emb_f32[id * 32 + c4]);
    float z0 = z_lds[row * Z_STRIDE + c4 + 0];
    float z1 = z_lds[row * Z_STRIDE + c4 + 1];
    float z2 = z_lds[row * Z_STRIDE + c4 + 2];
    float z3 = z_lds[row * Z_STRIDE + c4 + 3];
    *reinterpret_cast<float4*>(&out[(size_t)(m0 + row) * 32 + c4]) = e;  // z_q == emb[idx]
    float d0 = e.x - z0, d1 = e.y - z1, d2 = e.z - z2, d3 = e.w - z3;
    float ls = d0 * d0 + d1 * d1 + d2 * d2 + d3 * d3;
    #pragma unroll
    for (int mask = 16; mask >= 1; mask >>= 1) ls += __shfl_xor(ls, mask, 32);
    if (lane == 0) pl_lds[w] = ls;
  }
  __syncthreads();
  if (tid == 0) {
    float tot = 0.f;
    #pragma unroll
    for (int i = 0; i < 8; ++i) tot += pl_lds[i];
    partials[blockIdx.x] = tot * (1.0f / 1048576.0f);  // BETA * mean contribution
  }
}

// ---------- deterministic loss reduction (fixed order) ----------
__global__ __launch_bounds__(256) void k_loss_reduce(const float* __restrict__ p,
                                                     float* __restrict__ loss_out) {
  __shared__ float s[256];
  float a = 0.f;
  for (int i = threadIdx.x; i < 1024; i += 256) a += p[i];
  s[threadIdx.x] = a;
  __syncthreads();
  for (int st = 128; st > 0; st >>= 1) {
    if (threadIdx.x < st) s[threadIdx.x] += s[threadIdx.x + st];
    __syncthreads();
  }
  if (threadIdx.x == 0) loss_out[0] = s[0];
}

extern "C" void kernel_launch(void* const* d_in, const int* in_sizes, int n_in,
                              void* d_out, int out_size, void* d_ws, size_t ws_size,
                              hipStream_t stream) {
  (void)in_sizes; (void)n_in; (void)out_size; (void)ws_size;
  const float* features = (const float*)d_in[0];
  const float* W1 = (const float*)d_in[1];
  const float* b1 = (const float*)d_in[2];
  const float* W2 = (const float*)d_in[3];
  const float* b2 = (const float*)d_in[4];
  const float* cb = (const float*)d_in[5];
  float* out = (float*)d_out;

  char* ws = (char*)d_ws;
  float*          emb_f32  = (float*)(ws + WS_EMB_F32);
  unsigned short* emb_bf16 = (unsigned short*)(ws + WS_EMB_BF16);
  unsigned short* W1T      = (unsigned short*)(ws + WS_W1T);
  unsigned short* W2T      = (unsigned short*)(ws + WS_W2T);
  float*          partials = (float*)(ws + WS_PART);

  k_transpose_bf16<<<(768 * 768) / 256, 256, 0, stream>>>(W1, W1T, 768, 768);
  k_transpose_bf16<<<(768 * 32) / 256, 256, 0, stream>>>(W2, W2T, 768, 32);
  k_norm_codebook<<<K_CODES / 8, 256, 0, stream>>>(cb, emb_f32, emb_bf16);
  k_vqkd_fused<<<M_TOTAL / M_BLK, 256, LDS_BYTES, stream>>>(
      features, b1, b2, W1T, W2T, emb_bf16, emb_f32, out, partials);
  k_loss_reduce<<<1, 256, 0, stream>>>(partials, out + ZQ_ELEMS);
}
